// Mamba2BlockPure_55722905699087
// MI455X (gfx1250) — compile-verified, matrix operand order checked by reading, not run on hardware
//
#include <hip/hip_runtime.h>
#include <math.h>

#define D_MODEL 1024
#define D_INNER 2048
#define NHEADS  32
#define HEADDIM 64
#define D_STATE 64
#define D_CONV  4
#define BATCH   2
#define SEQ_LEN 1024
#define NROWS   (BATCH * SEQ_LEN)   // 2048 token rows

typedef __attribute__((ext_vector_type(16))) __bf16 v16bf;
typedef __attribute__((ext_vector_type(8)))  float  v8f;

// ---------------------------------------------------------------------------
// f32 -> bf16 (round to nearest even), stored as raw u16
// ---------------------------------------------------------------------------
__device__ __forceinline__ unsigned short f2bf(float f) {
    unsigned int u = __float_as_uint(f);
    unsigned int r = (u + 0x7FFFu + ((u >> 16) & 1u)) >> 16;
    return (unsigned short)r;
}

__global__ void k_f32_to_bf16(const float* __restrict__ in,
                              unsigned short* __restrict__ out, int n) {
    int i = blockIdx.x * blockDim.x + threadIdx.x;
    if (i < n) out[i] = f2bf(in[i]);
}

// ---------------------------------------------------------------------------
// WMMA bf16 GEMM:  Y[M,N] = X[M,K] * W[N,K]^T   (X,W bf16 row-major, Y f32)
// Block: 256 threads = 8 wave32, tile 128(M) x 64(N), K chunked by 32.
// Tiles staged global->LDS with GLOBAL_LOAD_ASYNC_TO_LDS_B128 (ASYNCcnt),
// double-buffered so the next chunk's HBM traffic overlaps current WMMAs.
// ---------------------------------------------------------------------------
#define TM 128
#define TN 64
#define TK 32
#define LDS_STRIDE 20   // uints per row: 16 data (32 bf16) + 4 pad (keeps 16B align)

__global__ __launch_bounds__(256)
void k_gemm_bf16(const unsigned short* __restrict__ X,
                 const unsigned short* __restrict__ W,
                 float* __restrict__ Y,
                 int M, int N, int K) {
    __shared__ unsigned int Xs[2][TM * LDS_STRIDE];
    __shared__ unsigned int Ws[2][TN * LDS_STRIDE];

    const int tid  = threadIdx.x;
    const int lane = tid & 31;
    const int wave = tid >> 5;
    const int wm   = wave & 3;   // 32-row slice within 128
    const int wn   = wave >> 2;  // 32-col slice within 64
    const int bm   = blockIdx.x * TM;
    const int bn   = blockIdx.y * TN;

    // Per-thread staging coordinates (3 async B128 ops per thread per chunk)
    const int xr0 = tid >> 2;            // X rows for it=0 (0..63) / it=1 (+64)
    const int xq  = tid & 3;             // 16B quad within the 32-wide K chunk
    int wr = bn + (tid >> 2);            // W row; clamp instead of zero-fill:
    if (wr >= N) wr = N - 1;             // dup data only feeds cols >= N (never stored)

    v8f acc[2][2];
    #pragma unroll
    for (int a = 0; a < 2; ++a)
        #pragma unroll
        for (int b = 0; b < 2; ++b)
            acc[a][b] = (v8f){0.f,0.f,0.f,0.f,0.f,0.f,0.f,0.f};

    const int frow = lane & 15;          // row (A) / col (B) inside 16-tile
    const int kb   = (lane >> 4) * 8;    // K base per ISA 16-bit A/B layout

    // issue one chunk's staging: 2 async b128 for X tile, 1 for W tile
    auto issue_chunk = [&](int k0, int buf) {
        #pragma unroll
        for (int it = 0; it < 2; ++it) {
            int r = xr0 + it * 64;
            unsigned ldsa = (unsigned)(size_t)(void*)&Xs[buf][r * LDS_STRIDE + xq * 4];
            const unsigned short* src = X + (size_t)(bm + r) * K + k0 + xq * 8;
            asm volatile("global_load_async_to_lds_b128 %0, %1, off"
                         :: "v"(ldsa), "v"(src) : "memory");
        }
        {
            int r = tid >> 2;
            unsigned ldsa = (unsigned)(size_t)(void*)&Ws[buf][r * LDS_STRIDE + xq * 4];
            const unsigned short* src = W + (size_t)wr * K + k0 + xq * 8;
            asm volatile("global_load_async_to_lds_b128 %0, %1, off"
                         :: "v"(ldsa), "v"(src) : "memory");
        }
    };

    const int nchunks = K / TK;
    issue_chunk(0, 0);

    for (int c = 0; c < nchunks; ++c) {
        const int buf = c & 1;
        if (c + 1 < nchunks) {
            issue_chunk((c + 1) * TK, buf ^ 1);
            // 6 outstanding, in-order completion: <=3 left means chunk c landed
            asm volatile("s_wait_asynccnt 0x3" ::: "memory");
        } else {
            asm volatile("s_wait_asynccnt 0x0" ::: "memory");
        }
        __syncthreads();   // everyone's chunk-c data visible in LDS

        union Frag { unsigned int u[8]; v16bf v; } afrag[2], bfrag[2];
        #pragma unroll
        for (int tm = 0; tm < 2; ++tm) {
            int row = wm * 32 + tm * 16 + frow;
            const unsigned int* base = &Xs[buf][row * LDS_STRIDE];
            #pragma unroll
            for (int vv = 0; vv < 8; ++vv) {
                int k = (vv < 4) ? (kb + 2 * vv) : (kb + 8 + 2 * vv);
                afrag[tm].u[vv] = base[k >> 1];
            }
        }
        #pragma unroll
        for (int tn = 0; tn < 2; ++tn) {
            int row = wn * 32 + tn * 16 + frow;
            const unsigned int* base = &Ws[buf][row * LDS_STRIDE];
            #pragma unroll
            for (int vv = 0; vv < 8; ++vv) {
                int k = (vv < 4) ? (kb + 2 * vv) : (kb + 8 + 2 * vv);
                bfrag[tn].u[vv] = base[k >> 1];
            }
        }

        #pragma unroll
        for (int tm = 0; tm < 2; ++tm)
            #pragma unroll
            for (int tn = 0; tn < 2; ++tn)
                acc[tm][tn] = __builtin_amdgcn_wmma_f32_16x16x32_bf16(
                    false, afrag[tm].v, false, bfrag[tn].v,
                    (short)0, acc[tm][tn], false, false);

        __syncthreads();   // done reading buf before it is re-targeted by async
    }

    // C/D layout: lane 0-15 -> N=lane, M=r ; lane 16-31 -> N=lane-16, M=8+r
    const int cn    = lane & 15;
    const int mhalf = (lane >> 4) * 8;
    #pragma unroll
    for (int tm = 0; tm < 2; ++tm) {
        #pragma unroll
        for (int tn = 0; tn < 2; ++tn) {
            int col = bn + wn * 32 + tn * 16 + cn;
            if (col >= N) continue;
            int rowb = bm + wm * 32 + tm * 16 + mhalf;
            #pragma unroll
            for (int r = 0; r < 8; ++r)
                Y[(size_t)(rowb + r) * N + col] = acc[tm][tn][r];
        }
    }
}

// ---------------------------------------------------------------------------
// Depthwise causal conv1d (k=4) + bias + SiLU on x-half of xz.
// ---------------------------------------------------------------------------
__global__ void k_conv_silu(const float* __restrict__ xz,
                            const float* __restrict__ conv_w,
                            const float* __restrict__ conv_b,
                            float* __restrict__ x_f32,
                            unsigned short* __restrict__ x_bf16) {
    int idx = blockIdx.x * blockDim.x + threadIdx.x;
    if (idx >= NROWS * D_INNER) return;
    int row = idx / D_INNER;
    int c   = idx - row * D_INNER;
    int b   = row / SEQ_LEN;
    int l   = row - b * SEQ_LEN;
    float acc = conv_b[c];
    #pragma unroll
    for (int j = 0; j < D_CONV; ++j) {
        int li = l - (D_CONV - 1) + j;
        if (li >= 0)
            acc += conv_w[c * D_CONV + j] *
                   xz[(size_t)(b * SEQ_LEN + li) * (2 * D_INNER) + c];
    }
    float s = acc / (1.0f + expf(-acc));   // SiLU
    x_f32[idx]  = s;
    x_bf16[idx] = f2bf(s);
}

// ---------------------------------------------------------------------------
// dt = softplus(dt_raw + dt_b)
// ---------------------------------------------------------------------------
__global__ void k_softplus_dt(const float* __restrict__ dtraw,
                              const float* __restrict__ dt_b,
                              float* __restrict__ dt) {
    int idx = blockIdx.x * blockDim.x + threadIdx.x;
    if (idx >= NROWS * NHEADS) return;
    int h = idx & (NHEADS - 1);
    float v = dtraw[idx] + dt_b[h];
    dt[idx] = (v > 20.0f) ? v : log1pf(expf(v));
}

// ---------------------------------------------------------------------------
// Selective scan: one block per (batch, head). 64x64 state in registers
// (16 elems/thread). Sequential over L; y reduced over N via 4-lane shuffles.
// Latency-bound -> prefetch next timestep's rows (global_prefetch_b8).
// ---------------------------------------------------------------------------
__global__ __launch_bounds__(256)
void k_scan(const float* __restrict__ x,   const float* __restrict__ dt,
            const float* __restrict__ Bm,  const float* __restrict__ Cm,
            const float* __restrict__ A_log, const float* __restrict__ Dv,
            float* __restrict__ y, float* __restrict__ hfin) {
    __shared__ float xs[HEADDIM], Bs[D_STATE], Cs[D_STATE], sc[2];
    const int bh  = blockIdx.x;
    const int b   = bh >> 5;
    const int hh  = bh & 31;
    const int tid = threadIdx.x;
    const int p   = tid >> 2;          // 0..63 : head-dim index
    const int n0  = (tid & 3) * 16;    // 16-wide N slice
    const float A  = -expf(A_log[hh]);
    const float Dh = Dv[hh];

    float h[16];
    #pragma unroll
    for (int i = 0; i < 16; ++i) h[i] = 0.0f;

    for (int t = 0; t < SEQ_LEN; ++t) {
        const size_t row = (size_t)b * SEQ_LEN + t;
        if (tid < 64)        xs[tid]       = x [row * D_INNER + hh * HEADDIM + tid];
        else if (tid < 128)  Bs[tid - 64]  = Bm[row * D_INNER + hh * D_STATE + (tid - 64)];
        else if (tid < 192)  Cs[tid - 128] = Cm[row * D_INNER + hh * D_STATE + (tid - 128)];
        else if (tid == 192) { float d = dt[row * NHEADS + hh]; sc[0] = expf(A * d); sc[1] = d; }
        // prefetch next timestep while this one is consumed
        if (t + 1 < SEQ_LEN) {
            const size_t nrow = row + 1;
            if (tid == 0)        __builtin_prefetch(&x [nrow * D_INNER + hh * HEADDIM], 0, 1);
            else if (tid == 64)  __builtin_prefetch(&Bm[nrow * D_INNER + hh * D_STATE], 0, 1);
            else if (tid == 128) __builtin_prefetch(&Cm[nrow * D_INNER + hh * D_STATE], 0, 1);
            else if (tid == 192) __builtin_prefetch(&dt[nrow * NHEADS + hh], 0, 1);
        }
        __syncthreads();

        const float Ab = sc[0], dv = sc[1], xp = xs[p];
        float part = 0.0f;
        #pragma unroll
        for (int i = 0; i < 16; ++i) {
            float hn = Ab * h[i] + (dv * Bs[n0 + i]) * xp;
            h[i] = hn;
            part += hn * Cs[n0 + i];
        }
        part += __shfl_xor(part, 1, 4);
        part += __shfl_xor(part, 2, 4);
        if ((tid & 3) == 0)
            y[row * D_INNER + hh * HEADDIM + p] = part + Dh * xp;
        __syncthreads();
    }
    #pragma unroll
    for (int i = 0; i < 16; ++i)
        hfin[((size_t)bh * HEADDIM + p) * D_STATE + n0 + i] = h[i];
}

// ---------------------------------------------------------------------------
// LayerNorm(y) * SiLU(z) -> bf16 for the output projection
// ---------------------------------------------------------------------------
__global__ __launch_bounds__(256)
void k_ln_gate(const float* __restrict__ y, const float* __restrict__ xz,
               const float* __restrict__ ln_w, const float* __restrict__ ln_b,
               unsigned short* __restrict__ out_bf) {
    __shared__ float red[256];
    const int row = blockIdx.x;
    const int tid = threadIdx.x;

    float s = 0.0f, ss = 0.0f;
    for (int c = tid; c < D_INNER; c += 256) {
        float v = y[(size_t)row * D_INNER + c];
        s += v; ss += v * v;
    }
    red[tid] = s; __syncthreads();
    for (int off = 128; off > 0; off >>= 1) {
        if (tid < off) red[tid] += red[tid + off];
        __syncthreads();
    }
    float mu = red[0] / (float)D_INNER;
    __syncthreads();
    red[tid] = ss; __syncthreads();
    for (int off = 128; off > 0; off >>= 1) {
        if (tid < off) red[tid] += red[tid + off];
        __syncthreads();
    }
    float var  = red[0] / (float)D_INNER - mu * mu;
    float rstd = rsqrtf(var + 1e-5f);

    for (int c = tid; c < D_INNER; c += 256) {
        float v = (y[(size_t)row * D_INNER + c] - mu) * rstd * ln_w[c] + ln_b[c];
        float z = xz[(size_t)row * (2 * D_INNER) + D_INNER + c];
        float g = z / (1.0f + expf(-z));   // SiLU(z)
        out_bf[(size_t)row * D_INNER + c] = f2bf(v * g);
    }
}

// ---------------------------------------------------------------------------
// Launcher
// ---------------------------------------------------------------------------
extern "C" void kernel_launch(void* const* d_in, const int* in_sizes, int n_in,
                              void* d_out, int out_size, void* d_ws, size_t ws_size,
                              hipStream_t stream) {
    (void)in_sizes; (void)n_in; (void)out_size; (void)ws_size;

    const float* hs    = (const float*)d_in[0];
    const float* Wip   = (const float*)d_in[1];
    const float* convw = (const float*)d_in[2];
    const float* convb = (const float*)d_in[3];
    const float* Alog  = (const float*)d_in[4];
    const float* Wdt   = (const float*)d_in[5];
    const float* dtb   = (const float*)d_in[6];
    const float* WB    = (const float*)d_in[7];
    const float* WC    = (const float*)d_in[8];
    const float* Dvec  = (const float*)d_in[9];
    const float* lnw   = (const float*)d_in[10];
    const float* lnb   = (const float*)d_in[11];
    const float* Wout  = (const float*)d_in[12];

    float* out  = (float*)d_out;                      // [B, L, D_MODEL]
    float* hfin = out + (size_t)NROWS * D_MODEL;      // [B, H, P, N]

    char*  ws  = (char*)d_ws;
    size_t off = 0;
    auto alloc = [&](size_t bytes) -> char* {
        char* p = ws + off;
        off += (bytes + 255) & ~(size_t)255;
        return p;
    };
    unsigned short* hs_bf   = (unsigned short*)alloc((size_t)NROWS * D_MODEL * 2);
    unsigned short* Wip_bf  = (unsigned short*)alloc((size_t)2 * D_INNER * D_MODEL * 2);
    unsigned short* Wdt_bf  = (unsigned short*)alloc((size_t)NHEADS * D_INNER * 2);
    unsigned short* WB_bf   = (unsigned short*)alloc((size_t)D_INNER * D_INNER * 2);
    unsigned short* WC_bf   = (unsigned short*)alloc((size_t)D_INNER * D_INNER * 2);
    unsigned short* Wout_bf = (unsigned short*)alloc((size_t)D_MODEL * D_INNER * 2);
    float*          xz      = (float*)alloc((size_t)NROWS * 2 * D_INNER * 4);
    float*          x_f     = (float*)alloc((size_t)NROWS * D_INNER * 4);
    unsigned short* x_bf    = (unsigned short*)alloc((size_t)NROWS * D_INNER * 2);
    float*          dtraw   = (float*)alloc((size_t)NROWS * NHEADS * 4);
    float*          dts     = (float*)alloc((size_t)NROWS * NHEADS * 4);
    float*          Bmat    = (float*)alloc((size_t)NROWS * D_INNER * 4);
    float*          Cmat    = (float*)alloc((size_t)NROWS * D_INNER * 4);
    float*          ybuf    = (float*)alloc((size_t)NROWS * D_INNER * 4);
    unsigned short* yn_bf   = (unsigned short*)alloc((size_t)NROWS * D_INNER * 2);

    dim3 blk(256);
    auto cvt = [&](const float* src, unsigned short* dst, int n) {
        k_f32_to_bf16<<<(n + 255) / 256, blk, 0, stream>>>(src, dst, n);
    };
    cvt(hs,   hs_bf,   NROWS * D_MODEL);
    cvt(Wip,  Wip_bf,  2 * D_INNER * D_MODEL);
    cvt(Wdt,  Wdt_bf,  NHEADS * D_INNER);
    cvt(WB,   WB_bf,   D_INNER * D_INNER);
    cvt(WC,   WC_bf,   D_INNER * D_INNER);
    cvt(Wout, Wout_bf, D_MODEL * D_INNER);

    // 1) xz = hs @ in_proj_w^T   [2048 x 4096], K=1024
    dim3 g1(NROWS / TM, (2 * D_INNER + TN - 1) / TN);
    k_gemm_bf16<<<g1, blk, 0, stream>>>(hs_bf, Wip_bf, xz, NROWS, 2 * D_INNER, D_MODEL);

    // 2) depthwise causal conv + SiLU
    k_conv_silu<<<(NROWS * D_INNER + 255) / 256, blk, 0, stream>>>(xz, convw, convb, x_f, x_bf);

    // 3) dt_raw = x @ dt_w^T  [2048 x 32], then softplus
    dim3 g2(NROWS / TM, (NHEADS + TN - 1) / TN);
    k_gemm_bf16<<<g2, blk, 0, stream>>>(x_bf, Wdt_bf, dtraw, NROWS, NHEADS, D_INNER);
    k_softplus_dt<<<(NROWS * NHEADS + 255) / 256, blk, 0, stream>>>(dtraw, dtb, dts);

    // 4) B and C projections [2048 x 2048], K=2048
    dim3 g3(NROWS / TM, D_INNER / TN);
    k_gemm_bf16<<<g3, blk, 0, stream>>>(x_bf, WB_bf, Bmat, NROWS, D_INNER, D_INNER);
    k_gemm_bf16<<<g3, blk, 0, stream>>>(x_bf, WC_bf, Cmat, NROWS, D_INNER, D_INNER);

    // 5) selective scan (writes y and h_final)
    k_scan<<<BATCH * NHEADS, blk, 0, stream>>>(x_f, dts, Bmat, Cmat, Alog, Dvec, ybuf, hfin);

    // 6) layernorm + SiLU(z) gate -> bf16
    k_ln_gate<<<NROWS, blk, 0, stream>>>(ybuf, xz, lnw, lnb, yn_bf);

    // 7) output = ynorm @ out_proj_w^T  [2048 x 1024], K=2048
    dim3 g4(NROWS / TM, D_MODEL / TN);
    k_gemm_bf16<<<g4, blk, 0, stream>>>(yn_bf, Wout_bf, out, NROWS, D_MODEL, D_INNER);
}
